// ResidualBlock_89069031785102
// MI455X (gfx1250) — matrix operand order repaired and checked by reading, not compiled, not run on hardware
//
#include <hip/hip_runtime.h>

typedef __attribute__((ext_vector_type(16))) __bf16 v16bf;
typedef __attribute__((ext_vector_type(8)))  float  v8f;

#define CIN     64
#define BB      4
#define HGT     64
#define WW      4096
#define HWs     (HGT * WW)          // 262144
#define CHW     (CIN * HWs)
#define WTILES  (WW / 16)           // 256
#define NTILES  (BB * HGT * WTILES) // 65536

__device__ __forceinline__ float sigm(float v) {
    return __builtin_amdgcn_rcpf(1.0f + __expf(-v));
}
__device__ __forceinline__ float tanh_fast(float v) {
    return 2.0f * __builtin_amdgcn_rcpf(1.0f + __expf(-2.0f * v)) - 1.0f;
}

#define WMMA_BF16(Afrag, Bfrag, Cfrag) \
    __builtin_amdgcn_wmma_f32_16x16x32_bf16(false, (Afrag), false, (Bfrag), \
                                            (short)0, (Cfrag), false, false)

__global__ __launch_bounds__(256) void
wavenet_block_kernel(const float* __restrict__ x,
                     const float* __restrict__ wl,   const float* __restrict__ bl,
                     const float* __restrict__ wr,   const float* __restrict__ br,
                     const float* __restrict__ wskp, const float* __restrict__ bskp,
                     const float* __restrict__ whid, const float* __restrict__ bhid,
                     float* __restrict__ out1, float* __restrict__ out2)
{
    // A fragments: [frag 0..31][lane 0..31][half 0..15] bf16, 32 KB
    __shared__ __align__(64) __bf16 sA[32 * 32 * 16];
    __shared__ __align__(64) float  sBias[4][64];

    // ---- Stage weights into LDS in WMMA A-fragment layout (16x32 bf16) ----
    // frag f = gemm*8 + mt*2 + kt ; A element (lane, hh):
    //   M = 16*mt + (lane&15)
    //   K = 32*kt + 16*(lane>>4) + hh
    for (int e = threadIdx.x; e < 32 * 32 * 16; e += 256) {
        int hh   = e & 15;
        int lane = (e >> 4) & 31;
        int f    = e >> 9;
        int gemm = f >> 3, mt = (f >> 1) & 3, kt = f & 1;
        int v = hh >> 1, wv = hh & 1;
        int M = 16 * mt + (lane & 15);
        int K = 32 * kt + 16 * (lane >> 4) + hh;
        float val;
        if (gemm == 0)      // conv_l: tap1 = current, tap0 = w-4
            val = (K < 32) ? wl[M * 64 + K * 2 + 1] : wl[M * 64 + (K - 32) * 2 + 0];
        else if (gemm == 1) // conv_r: tap0 = current, tap1 = w+4
            val = (K < 32) ? wr[M * 64 + K * 2 + 0] : wr[M * 64 + (K - 32) * 2 + 1];
        else if (gemm == 2)
            val = wskp[M * 64 + K];
        else
            val = whid[M * 64 + K];
        sA[e] = (__bf16)val;
    }
    for (int e = threadIdx.x; e < 4 * 64; e += 256) {
        int vsel = e >> 6, c = e & 63;
        const float* bp = (vsel == 0) ? bl : (vsel == 1) ? br : (vsel == 2) ? bskp : bhid;
        sBias[vsel][c] = bp[c];
    }
    __syncthreads();

    const int lane = threadIdx.x & 31;
    const int hL   = lane >> 4;   // lane half
    const int p    = lane & 15;   // pixel within tile (N index)
    const int gw   = blockIdx.x * (blockDim.x >> 5) + (threadIdx.x >> 5);
    const int nw   = gridDim.x * (blockDim.x >> 5);

    for (int t = gw; t < NTILES; t += nw) {
        // Opaque zero, re-defined every iteration: forces all LDS reads of
        // weights/bias to stay inside the loop (no LICM -> no register hoard
        // -> no scratch spills). One SALU op per tile.
        int zoff;
        asm volatile("s_mov_b32 %0, 0" : "=s"(zoff));
        const __bf16* sAz = sA + zoff;
        const float*  sBz = &sBias[0][0] + zoff;   // sBias[v][c] = sBz[v*64+c]

        int tb  = t / (HGT * WTILES);
        int rem = t - tb * (HGT * WTILES);
        int th  = rem / WTILES;
        int tw  = rem - th * WTILES;
        int w0  = tw << 4;
        long long base = (long long)tb * CHW + (long long)th * WW + w0;

        bool okp = (w0 + p) >= 4;          // left (causal) tap valid
        bool okn = (w0 + p) <= WW - 5;     // right tap valid

        // ======== Phase A: left conv (channels 0..31, taps w / w-4) ========
        v16bf Bl0, Bl1;
        #pragma unroll
        for (int hh = 0; hh < 16; ++hh) {
            int c = 16 * hL + hh;                               // 0..31
            long long oc = base + (long long)c * HWs + p;
            Bl0[hh] = (__bf16)x[oc];
            Bl1[hh] = (__bf16)(okp ? x[oc - 4] : 0.0f);
        }
        v8f accL[4];
        #pragma unroll
        for (int mt = 0; mt < 4; ++mt) {
            accL[mt] = *(const v8f*)&sBz[0 * 64 + 16 * mt + 8 * hL];
            v16bf a;
            a = *(const v16bf*)(sAz + ((0 * 8 + mt * 2 + 0) * 512 + lane * 16));
            accL[mt] = WMMA_BF16(a, Bl0, accL[mt]);
            a = *(const v16bf*)(sAz + ((0 * 8 + mt * 2 + 1) * 512 + lane * 16));
            accL[mt] = WMMA_BF16(a, Bl1, accL[mt]);
        }

        // ======== Phase B: right conv (channels 32..63, taps w / w+4) ======
        v16bf Br0, Br1;
        #pragma unroll
        for (int hh = 0; hh < 16; ++hh) {
            int c = 16 * hL + hh;
            long long oc = base + (long long)(32 + c) * HWs + p;
            Br0[hh] = (__bf16)x[oc];
            Br1[hh] = (__bf16)(okn ? x[oc + 4] : 0.0f);
        }
        v8f accR[4];
        #pragma unroll
        for (int mt = 0; mt < 4; ++mt) {
            accR[mt] = *(const v8f*)&sBz[1 * 64 + 16 * mt + 8 * hL];
            v16bf a;
            a = *(const v16bf*)(sAz + ((1 * 8 + mt * 2 + 0) * 512 + lane * 16));
            accR[mt] = WMMA_BF16(a, Br0, accR[mt]);
            a = *(const v16bf*)(sAz + ((1 * 8 + mt * 2 + 1) * 512 + lane * 16));
            accR[mt] = WMMA_BF16(a, Br1, accR[mt]);
        }

        // ======== Gated activation (elementwise in fragment registers) =====
        v8f hC[4];
        #pragma unroll
        for (int r = 0; r < 8; ++r) {
            hC[0][r] = tanh_fast(accL[0][r]) * sigm(accL[2][r]);
            hC[1][r] = tanh_fast(accL[1][r]) * sigm(accL[3][r]);
            hC[2][r] = tanh_fast(accR[0][r]) * sigm(accR[2][r]);
            hC[3][r] = tanh_fast(accR[1][r]) * sigm(accR[3][r]);
        }

        // ==== C-layout -> B-layout for h: one SWAPX16 per needed tile ======
        v16bf Bh[2];
        #pragma unroll
        for (int kt = 0; kt < 2; ++kt) {
            v8f ts    = hL ? hC[2 * kt + 1] : hC[2 * kt + 0];  // own-half values
            v8f tsel2 = hL ? hC[2 * kt + 0] : hC[2 * kt + 1];  // what other half needs
            v8f tsw;
            #pragma unroll
            for (int r = 0; r < 8; ++r)
                tsw[r] = __int_as_float(
                    __builtin_amdgcn_ds_swizzle(__float_as_int(tsel2[r]), 0x401F));
            #pragma unroll
            for (int hh = 0; hh < 16; ++hh) {
                int r = hh & 7;
                float sv = (((hh >> 3) & 1) == hL) ? ts[r] : tsw[r];
                Bh[kt][hh] = (__bf16)sv;
            }
        }

        // ======== Phase C: skip output  o = Wskp*h + bskp ==================
        {
            v8f accO[4];
            #pragma unroll
            for (int mt = 0; mt < 4; ++mt) {
                accO[mt] = *(const v8f*)&sBz[2 * 64 + 16 * mt + 8 * hL];
                v16bf a;
                a = *(const v16bf*)(sAz + ((2 * 8 + mt * 2 + 0) * 512 + lane * 16));
                accO[mt] = WMMA_BF16(a, Bh[0], accO[mt]);
                a = *(const v16bf*)(sAz + ((2 * 8 + mt * 2 + 1) * 512 + lane * 16));
                accO[mt] = WMMA_BF16(a, Bh[1], accO[mt]);
            }
            #pragma unroll
            for (int mt = 0; mt < 4; ++mt)
                #pragma unroll
                for (int r = 0; r < 8; ++r) {
                    int c = 16 * mt + 8 * hL + r;
                    out2[base + (long long)c * HWs + p] = accO[mt][r];
                }
        }

        // ======== Phase D: residual  out1 = x + (Whid*h + bhid) ============
        {
            v8f accH[4];
            #pragma unroll
            for (int mt = 0; mt < 4; ++mt) {
                accH[mt] = *(const v8f*)&sBz[3 * 64 + 16 * mt + 8 * hL];
                v16bf a;
                a = *(const v16bf*)(sAz + ((3 * 8 + mt * 2 + 0) * 512 + lane * 16));
                accH[mt] = WMMA_BF16(a, Bh[0], accH[mt]);
                a = *(const v16bf*)(sAz + ((3 * 8 + mt * 2 + 1) * 512 + lane * 16));
                accH[mt] = WMMA_BF16(a, Bh[1], accH[mt]);
            }
            #pragma unroll
            for (int mt = 0; mt < 4; ++mt)
                #pragma unroll
                for (int r = 0; r < 8; ++r) {
                    int c = 16 * mt + 8 * hL + r;
                    long long off = base + (long long)c * HWs + p;
                    out1[off] = x[off] + accH[mt][r];  // full-precision residual
                }
        }
    }
}

extern "C" void kernel_launch(void* const* d_in, const int* in_sizes, int n_in,
                              void* d_out, int out_size, void* d_ws, size_t ws_size,
                              hipStream_t stream) {
    const float* x    = (const float*)d_in[0];
    const float* wl   = (const float*)d_in[1];
    const float* bl   = (const float*)d_in[2];
    const float* wr   = (const float*)d_in[3];
    const float* br   = (const float*)d_in[4];
    const float* wskp = (const float*)d_in[5];
    const float* bskp = (const float*)d_in[6];
    const float* whid = (const float*)d_in[7];
    const float* bhid = (const float*)d_in[8];
    float* out1 = (float*)d_out;
    float* out2 = out1 + (size_t)BB * CHW;

    dim3 grid(1024), block(256);  // 8192 waves, 8 tiles each (65536 tiles)
    wavenet_block_kernel<<<grid, block, 0, stream>>>(
        x, wl, bl, wr, br, wskp, bskp, whid, bhid, out1, out2);
}
